// CharDecoder_48911087567199
// MI455X (gfx1250) — matrix operand order repaired and checked by reading, not compiled, not run on hardware
//
#include <hip/hip_runtime.h>

// ---------------- problem constants ----------------
constexpr int kB = 4096;
constexpr int kT = 20;
constexpr int kS = 20;
constexpr int kH = 512;
constexpr int kD = 512;
constexpr int kG = 3 * kH;   // 1536

// ---------------- WMMA vector types ----------------
typedef __attribute__((ext_vector_type(16))) __bf16 v16bf;
typedef __attribute__((ext_vector_type(8)))  __bf16 bf16x8;
typedef __attribute__((ext_vector_type(8)))  float  v8f;
typedef int v4i_gcc __attribute__((__vector_size__(16)));   // matches builtin param

union AFrag { v16bf v; bf16x8 h[2]; };

#ifndef __has_builtin
#define __has_builtin(x) 0
#endif
#if __has_builtin(__builtin_amdgcn_global_load_async_to_lds_b128)
#define HAVE_ASYNC_LDS 1
#else
#define HAVE_ASYNC_LDS 0
#endif

// ---------------- prep kernels ----------------

// f32 [R,C] row-major -> bf16 [C,R] row-major (i.e. transpose + convert)
__global__ void k_transpose_bf16(const float* __restrict__ src,
                                 __bf16* __restrict__ dst, int R, int C) {
  int idx = blockIdx.x * blockDim.x + threadIdx.x;
  if (idx >= R * C) return;
  int r = idx / C, c = idx % C;
  dst[c * R + r] = (__bf16)src[idx];
}

__global__ void k_f32_to_bf16(const float* __restrict__ src,
                              __bf16* __restrict__ dst, int n) {
  int idx = blockIdx.x * blockDim.x + threadIdx.x;
  if (idx < n) dst[idx] = (__bf16)src[idx];
}

__global__ void k_init_h(const float* __restrict__ cond,
                         float* __restrict__ h, __bf16* __restrict__ hb, int n) {
  int idx = blockIdx.x * blockDim.x + threadIdx.x;
  if (idx < n) { float v = cond[idx]; h[idx] = v; hb[idx] = (__bf16)v; }
}

// masked embedding gather for timestep t -> bf16 [B,D]
__global__ void k_embed_step(const int* __restrict__ tok,
                             const int* __restrict__ wlen,
                             const float* __restrict__ emb,
                             __bf16* __restrict__ embT, int t) {
  int idx = blockIdx.x * blockDim.x + threadIdx.x;
  if (idx >= kB * kD) return;
  int b = idx >> 9, d = idx & (kD - 1);
  int len = wlen[b]; if (len < 1) len = 1;
  float v = 0.0f;
  if (t < len) {
    int tk = tok[b * kT + t];
    v = emb[(size_t)tk * kD + d];
  }
  embT[idx] = (__bf16)v;
}

// cat = [emb_t, h]  -> bf16 [B, 2H]
__global__ void k_pack_cat(const __bf16* __restrict__ embT,
                           const __bf16* __restrict__ hb,
                           __bf16* __restrict__ cat) {
  int idx = blockIdx.x * blockDim.x + threadIdx.x;
  if (idx >= kB * 2 * kH) return;
  int b = idx >> 10, c = idx & (2 * kH - 1);
  cat[idx] = (c < kH) ? embT[b * kH + c] : hb[b * kH + (c - kH)];
}

// ---------------- bf16 WMMA GEMM ----------------
// C[M,N] (f32) = A[M,K] (bf16, row-major) * Bm[K,N] (bf16, row-major) + bias[N]
// block: 256 threads = 8 waves (2 M-waves x 4 N-waves).
// Wave tile: (MI*16) x (NJ*16). Block tile: (32*MI) x (64*NJ).
// Requires M % (32*MI) == 0, N % (64*NJ) == 0, K % 32 == 0.
// No divergence anywhere -> EXEC stays all-ones as WMMA requires.
template <int MI, int NJ>
__global__ __launch_bounds__(256)
void gemm_bf16_wmma(const __bf16* __restrict__ A, const __bf16* __restrict__ Bm,
                    const float* __restrict__ bias, float* __restrict__ C,
                    int M, int N, int K) {
  const int lane = threadIdx.x & 31;
  const int wave = threadIdx.x >> 5;
  const int wm = wave & 1;        // 2 waves along M
  const int wn = wave >> 1;       // 4 waves along N
  const int l15 = lane & 15;
  const int hi  = lane >> 4;
  const int m0 = blockIdx.x * (32 * MI) + wm * (16 * MI);
  const int n0 = blockIdx.y * (64 * NJ) + wn * (16 * NJ);

  v8f acc[MI][NJ];
#pragma unroll
  for (int i = 0; i < MI; ++i)
#pragma unroll
    for (int j = 0; j < NJ; ++j)
      acc[i][j] = (v8f){0.f, 0.f, 0.f, 0.f, 0.f, 0.f, 0.f, 0.f};

  for (int kt = 0; kt < K; kt += 32) {
    // A fragments: 16-bit A 16x32 layout (ISA 7.12.2):
    // lane l -> row (l&15); VGPR0..3 hold K = (l>>4)*8 .. +7, VGPR4..7 hold +16.
    AFrag a[MI];
#pragma unroll
    for (int i = 0; i < MI; ++i) {
      const __bf16* ap = A + (size_t)(m0 + i * 16 + l15) * K + kt + hi * 8;
      a[i].h[0] = *(const bf16x8*)(ap);
      a[i].h[1] = *(const bf16x8*)(ap + 16);
    }
    // B fragments: 32x16, lane -> column N = (l&15); elements 0..7 hold
    // K = (l>>4)*8 .. +7, elements 8..15 hold +16 (same order as A).
#pragma unroll
    for (int j = 0; j < NJ; ++j) {
      v16bf bfrag;
      const __bf16* bp = Bm + (size_t)(kt + hi * 8) * N + n0 + j * 16 + l15;
#pragma unroll
      for (int s = 0; s < 8; ++s) {
        bfrag[s] = bp[(size_t)s * N];
        bfrag[s + 8] = bp[(size_t)(16 + s) * N];
      }
#pragma unroll
      for (int i = 0; i < MI; ++i)
        acc[i][j] = __builtin_amdgcn_wmma_f32_16x16x32_bf16(
            false, a[i].v, false, bfrag, (short)0, acc[i][j], false, false);
    }
  }

  // Epilogue: C/D layout: VGPR r -> M = r + 8*hi, N = l15.
#pragma unroll
  for (int i = 0; i < MI; ++i) {
#pragma unroll
    for (int j = 0; j < NJ; ++j) {
      int col = n0 + j * 16 + l15;
      float bv = bias ? bias[col] : 0.0f;
#pragma unroll
      for (int r = 0; r < 8; ++r) {
        int row = m0 + i * 16 + hi * 8 + r;
        C[(size_t)row * N + col] = acc[i][j][r] + bv;
      }
    }
  }
}

// ---------------- attention scores + softmax ----------------
// scores[b,s] = sum_h enc_bf16[b,s,h] * energy[b,h]; softmax over s -> attn out
__global__ __launch_bounds__(256)
void k_attn(const __bf16* __restrict__ encb, const float* __restrict__ energy,
            float* __restrict__ attn_out, int t) {
  __shared__ alignas(16) float es[kH];
  __shared__ float ss[32];
  const int b = blockIdx.x;
  const int tid = threadIdx.x;

  // Stage energy[b,:] (2 KB) into LDS. Prefer the CDNA5 async global->LDS path.
#if HAVE_ASYNC_LDS
  if (tid < kH / 4) {
    __builtin_amdgcn_global_load_async_to_lds_b128(
        (__attribute__((address_space(1))) v4i_gcc*)(const void*)(energy + (size_t)b * kH + tid * 4),
        (__attribute__((address_space(3))) v4i_gcc*)(es + tid * 4), 0, 0);
  }
#if __has_builtin(__builtin_amdgcn_s_wait_asynccnt)
  __builtin_amdgcn_s_wait_asynccnt(0);
#else
  asm volatile("s_wait_asynccnt 0x0" ::: "memory");
#endif
  __syncthreads();
#else
  for (int i = tid; i < kH; i += 256) es[i] = energy[(size_t)b * kH + i];
  __syncthreads();
#endif

  if (tid < kS * 8) {                 // 8 lanes cooperate per s (contiguous lanes)
    int s = tid >> 3, part = tid & 7;
    const __bf16* ep = encb + ((size_t)b * kS + s) * kH;
    float sum = 0.0f;
    int h0 = part * (kH / 8);
    for (int h = h0; h < h0 + kH / 8; ++h) sum += (float)ep[h] * es[h];
    sum += __shfl_xor(sum, 1, 32);
    sum += __shfl_xor(sum, 2, 32);
    sum += __shfl_xor(sum, 4, 32);
    if (part == 0) ss[s] = sum;
  }
  __syncthreads();
  if (tid < 32) {                     // one wave does the 20-wide softmax
    float v = (tid < kS) ? ss[tid] : -3.0e38f;
    float mx = v;
    for (int o = 16; o; o >>= 1) mx = fmaxf(mx, __shfl_xor(mx, o, 32));
    float e = (tid < kS) ? __expf(v - mx) : 0.0f;
    float sm = e;
    for (int o = 16; o; o >>= 1) sm += __shfl_xor(sm, o, 32);
    if (tid < kS) attn_out[((size_t)b * kT + t) * kS + tid] = e / sm;
  }
}

// ---------------- GRU cell elementwise ----------------
__global__ void k_gru(const float* __restrict__ xg, const float* __restrict__ hg,
                      float* __restrict__ h, __bf16* __restrict__ hb,
                      float* __restrict__ out_h, int t) {
  int idx = blockIdx.x * blockDim.x + threadIdx.x;
  if (idx >= kB * kH) return;
  int b = idx >> 9, c = idx & (kH - 1);
  const float* x  = xg + (size_t)b * kG;
  const float* hh = hg + (size_t)b * kG;
  float xr = x[c],  xz = x[kH + c],  xn = x[2 * kH + c];
  float hr = hh[c], hz = hh[kH + c], hn = hh[2 * kH + c];
  float hp = h[idx];
  float r = 1.0f / (1.0f + __expf(-(xr + hr)));
  float z = 1.0f / (1.0f + __expf(-(xz + hz)));
  float n = tanhf(xn + r * hn);
  float hnew = (1.0f - z) * n + z * hp;
  h[idx] = hnew;
  hb[idx] = (__bf16)hnew;
  out_h[((size_t)b * kT + t) * kH + c] = hnew;
}

// ---------------- launcher ----------------
extern "C" void kernel_launch(void* const* d_in, const int* in_sizes, int n_in,
                              void* d_out, int out_size, void* d_ws, size_t ws_size,
                              hipStream_t stream) {
  (void)in_sizes; (void)n_in; (void)out_size; (void)ws_size;
  const int*   tok   = (const int*)d_in[0];    // output          [B,T]
  const int*   wlen  = (const int*)d_in[1];    // output_word_len [B]
  const float* cond  = (const float*)d_in[2];  // conditioning    [1,B,H]
  const float* enc   = (const float*)d_in[3];  // encoder         [B,S,H]
  const float* emb   = (const float*)d_in[4];  // emb_table       [V,D]
  const float* Wih   = (const float*)d_in[5];  // [3H,D]
  const float* Whh   = (const float*)d_in[6];  // [3H,H]
  const float* bih   = (const float*)d_in[7];  // [3H]
  const float* bhh   = (const float*)d_in[8];  // [3H]
  const float* attnW = (const float*)d_in[9];  // [H,2H]
  const float* attnb = (const float*)d_in[10]; // [H]

  float* out_h = (float*)d_out;                       // [B,T,H]
  float* out_a = out_h + (size_t)kB * kT * kH;        // [B,T,S]

  // workspace carve-up (256B aligned slabs), ~180 MB total
  char* p = (char*)d_ws;
  auto carve = [&](size_t bytes) -> void* {
    void* r = (void*)p;
    p += (bytes + 255) & ~(size_t)255;
    return r;
  };
  __bf16* WihT   = (__bf16*)carve((size_t)kD * kG * 2);        // [D,3H]
  __bf16* WhhT   = (__bf16*)carve((size_t)kH * kG * 2);        // [H,3H]
  __bf16* attnWT = (__bf16*)carve((size_t)2 * kH * kH * 2);    // [2H,H]
  __bf16* encb   = (__bf16*)carve((size_t)kB * kS * kH * 2);   // [B,S,H]
  float*  hF     = (float*) carve((size_t)kB * kH * 4);        // [B,H]
  __bf16* hB     = (__bf16*)carve((size_t)kB * kH * 2);        // [B,H]
  __bf16* embT   = (__bf16*)carve((size_t)kB * kD * 2);        // [B,D]
  __bf16* cat    = (__bf16*)carve((size_t)kB * 2 * kH * 2);    // [B,2H]
  float*  xg     = (float*) carve((size_t)kB * kG * 4);        // [B,3H]
  float*  hg     = (float*) carve((size_t)kB * kG * 4);        // [B,3H]
  float*  energy = (float*) carve((size_t)kB * kH * 4);        // [B,H]

  auto nb = [](size_t n) { return dim3((unsigned)((n + 255) / 256)); };

  // one-time prep (re-done every call: deterministic, inputs never mutated)
  k_transpose_bf16<<<nb((size_t)kG * kD), 256, 0, stream>>>(Wih, WihT, kG, kD);
  k_transpose_bf16<<<nb((size_t)kG * kH), 256, 0, stream>>>(Whh, WhhT, kG, kH);
  k_transpose_bf16<<<nb((size_t)kH * 2 * kH), 256, 0, stream>>>(attnW, attnWT, kH, 2 * kH);
  k_f32_to_bf16<<<nb((size_t)kB * kS * kH), 256, 0, stream>>>(enc, encb, kB * kS * kH);
  k_init_h<<<nb((size_t)kB * kH), 256, 0, stream>>>(cond, hF, hB, kB * kH);

  for (int t = 0; t < kT; ++t) {
    k_embed_step<<<nb((size_t)kB * kD), 256, 0, stream>>>(tok, wlen, emb, embT, t);
    k_pack_cat<<<nb((size_t)kB * 2 * kH), 256, 0, stream>>>(embT, hB, cat);
    // energy = cat[B,2H] x attnW^T[2H,H] + attn_b
    // small-N GEMM: 32x64 wave tile keeps 1024 waves in flight
    gemm_bf16_wmma<2, 4><<<dim3(kB / 64, kH / 256), 256, 0, stream>>>(
        cat, attnWT, attnb, energy, kB, kH, 2 * kH);
    k_attn<<<dim3(kB), 256, 0, stream>>>(encb, energy, out_a, t);
    // x_g = emb_t[B,D] x Wih^T[D,3H] + b_ih
    // big GEMMs: 64x64 wave tile -> 1.0 b128 load per WMMA
    gemm_bf16_wmma<4, 4><<<dim3(kB / 128, kG / 256), 256, 0, stream>>>(
        embT, WihT, bih, xg, kB, kG, kD);
    // h_g = h[B,H] x Whh^T[H,3H] + b_hh
    gemm_bf16_wmma<4, 4><<<dim3(kB / 128, kG / 256), 256, 0, stream>>>(
        hB, WhhT, bhh, hg, kB, kG, kH);
    k_gru<<<nb((size_t)kB * kH), 256, 0, stream>>>(xg, hg, hF, hB, out_h, t);
  }
}
